// Attention_nonlinear_29231547416586
// MI455X (gfx1250) — matrix operand order repaired and checked by reading, not compiled, not run on hardware
//
#include <hip/hip_runtime.h>
#include <hip/hip_bf16.h>

// CDNA5 / gfx1250, wave32. Main GEMM: v_wmma_f32_16x16x32_bf16, 2x2 tiled.

typedef __attribute__((ext_vector_type(16))) __bf16 v16bf;
typedef __attribute__((ext_vector_type(8)))  float  v8f;

constexpr int Bsz  = 32;    // batch
constexpr int Cin  = 256;   // C_IN (K of main GEMM)
constexpr int Cout = 512;   // C_OUT (K of gate GEMM)
constexpr int Adim = 256;   // hidden A
constexpr int HW   = 4096;  // 64*64 spatial
constexpr int PT   = 32;    // positions per workgroup (2 WMMA n-tiles)

// ---------------------------------------------------------------------------
// Kernel 0: pre-swizzle Wx (fp32, row-major [A, Cin]) into bf16 laid out in
// the per-lane A-fragment order used by the main loop:
//   wxs[((aTile*8 + ks)*32 + lane)*16 + i]
// so the main loop loads one contiguous 32B chunk per lane per k-step.
// A order used here (contiguous, not AMD's table order): lane half hl holds K = hl*16+0..15 in elems 0..15.
// ---------------------------------------------------------------------------
__global__ void wx_swizzle_kernel(const float* __restrict__ Wx,
                                  __bf16* __restrict__ wxs) {
  const int idx  = blockIdx.x * 256 + threadIdx.x;   // 65536 total
  const int i    = idx & 15;
  const int lane = (idx >> 4) & 31;
  const int ks   = (idx >> 9) & 7;
  const int aT   = idx >> 12;
  const int row  = lane & 15;
  const int hl   = lane >> 4;
  const int k    = ks * 32 + hl * 16 + i;
  wxs[idx] = (__bf16)Wx[(size_t)(aT * 16 + row) * Cin + k];
}

// ---------------------------------------------------------------------------
// Kernel 1: ggbx[b,a] = g[b,:] . Wg[a,:] + bg[a] + bx[a]   (tiny, VALU)
// ---------------------------------------------------------------------------
__global__ void gate_kernel(const float* __restrict__ g, const float* __restrict__ Wg,
                            const float* __restrict__ bg, const float* __restrict__ bx,
                            float* __restrict__ ggbx) {
  const int b = blockIdx.x;
  const int a = threadIdx.x;
  const float* gr = g  + (size_t)b * Cout;
  const float* wr = Wg + (size_t)a * Cout;
  float s = bg[a] + bx[a];
  for (int k = 0; k < Cout; ++k) s = fmaf(gr[k], wr[k], s);
  ggbx[b * Adim + a] = s;
}

// ---------------------------------------------------------------------------
// Kernel 2: main fused WMMA kernel. WG = (b, 32-position tile), 8 waves.
//   nx^T[a,pos] = sum_c Wx[a,c] * x[b,c,pos]  (bf16 WMMA, f32 acc)
//   att_raw[b,pos] = sum_a relu(nx + gg + bx) * Wf[a] + bf0   (fused)
// Each wave: 2 a-tiles x 2 pos-tiles -> 4 accumulators, 4 WMMA per k-step.
// ---------------------------------------------------------------------------
__global__ void __launch_bounds__(256)
att_gemm_kernel(const float* __restrict__ x, const __bf16* __restrict__ wxs,
                const float* __restrict__ Wf, const float* __restrict__ bf,
                const float* __restrict__ ggbx, float* __restrict__ att_raw) {
  __shared__ __align__(32) __bf16 ldsBT[PT * Cin];   // [pos][c] bf16, 16 KB
  __shared__ float ldsPart[2 * 256];                 // per pos-tile partials

  const int b       = blockIdx.y;
  const int posBase = blockIdx.x * PT;
  const int tid     = threadIdx.x;
  const int lane    = tid & 31;
  const int wave    = tid >> 5;

  // --- stage x[b, :, posBase:+32] -> bf16, transposed to [pos][c] ---
  // Each thread: 4 consecutive channels x 8 positions. Packed cvt + b64 DS
  // stores (4 bf16 per store, 8-byte aligned, no shared-DWORD writes).
  {
    const int c0 = (tid & 63) * 4;        // channel quad
    const int qh = tid >> 6;              // position-octet 0..3
    const float* xr = x + ((size_t)b * Cin + c0) * HW + posBase + qh * 8;
    float4 r0a = ((const float4*)(xr + 0 * HW))[0];
    float4 r0b = ((const float4*)(xr + 0 * HW))[1];
    float4 r1a = ((const float4*)(xr + 1 * HW))[0];
    float4 r1b = ((const float4*)(xr + 1 * HW))[1];
    float4 r2a = ((const float4*)(xr + 2 * HW))[0];
    float4 r2b = ((const float4*)(xr + 2 * HW))[1];
    float4 r3a = ((const float4*)(xr + 3 * HW))[0];
    float4 r3b = ((const float4*)(xr + 3 * HW))[1];
    const float rows[4][8] = {
        {r0a.x, r0a.y, r0a.z, r0a.w, r0b.x, r0b.y, r0b.z, r0b.w},
        {r1a.x, r1a.y, r1a.z, r1a.w, r1b.x, r1b.y, r1b.z, r1b.w},
        {r2a.x, r2a.y, r2a.z, r2a.w, r2b.x, r2b.y, r2b.z, r2b.w},
        {r3a.x, r3a.y, r3a.z, r3a.w, r3b.x, r3b.y, r3b.z, r3b.w}};
    #pragma unroll
    for (int p = 0; p < 8; ++p) {
      union { __bf16 h[4]; uint2 u; } pk;
      pk.h[0] = (__bf16)rows[0][p];
      pk.h[1] = (__bf16)rows[1][p];
      pk.h[2] = (__bf16)rows[2][p];
      pk.h[3] = (__bf16)rows[3][p];
      *reinterpret_cast<uint2*>(&ldsBT[(qh * 8 + p) * Cin + c0]) = pk.u;
    }
  }
  __syncthreads();

  const int hl  = lane >> 4;          // lane half
  const int aT0 = wave * 2;           // wave owns a-tiles 2w, 2w+1
  const int aT1 = wave * 2 + 1;

  v8f acc00 = {}, acc01 = {}, acc10 = {}, acc11 = {};

  // Pre-swizzled A streams: 32B per k-step per lane, contiguous.
  const __bf16* aptr0 = wxs + ((size_t)(aT0 * 8) * 32 + lane) * 16;
  const __bf16* aptr1 = wxs + ((size_t)(aT1 * 8) * 32 + lane) * 16;
  // B fragments from LDS: column n = lane&15; lanes 0-15 hold K 0..15,
  // lanes 16-31 hold K 16..31 of the step (sequential within VGPRs).
  const __bf16* bsrc = &ldsBT[(lane & 15) * Cin + hl * 16];

  #pragma unroll
  for (int ks = 0; ks < 8; ++ks) {
    const v16bf a0 = *reinterpret_cast<const v16bf*>(aptr0 + ks * 512);
    const v16bf a1 = *reinterpret_cast<const v16bf*>(aptr1 + ks * 512);
    const v16bf b0 = *reinterpret_cast<const v16bf*>(bsrc + ks * 32);
    const v16bf b1 = *reinterpret_cast<const v16bf*>(bsrc + 16 * Cin + ks * 32);
    acc00 = __builtin_amdgcn_wmma_f32_16x16x32_bf16(false, a0, false, b0,
                                                    (short)0, acc00, false, false);
    acc01 = __builtin_amdgcn_wmma_f32_16x16x32_bf16(false, a0, false, b1,
                                                    (short)0, acc01, false, false);
    acc10 = __builtin_amdgcn_wmma_f32_16x16x32_bf16(false, a1, false, b0,
                                                    (short)0, acc10, false, false);
    acc11 = __builtin_amdgcn_wmma_f32_16x16x32_bf16(false, a1, false, b1,
                                                    (short)0, acc11, false, false);
  }

  // --- fused epilogue: relu(nx + gg + bx) . Wf  from D fragments ---
  // D layout: N(pos) = lane&15 ; row M = v + 8*(lane>>4).
  float c0 = 0.f, c1 = 0.f;
  const float* gb = ggbx + (size_t)b * Adim;
  #pragma unroll
  for (int v = 0; v < 8; ++v) {
    const int a0i = aT0 * 16 + v + 8 * hl;
    const int a1i = aT1 * 16 + v + 8 * hl;
    const float w0 = Wf[a0i], g0 = gb[a0i];
    const float w1 = Wf[a1i], g1 = gb[a1i];
    c0 += fmaxf(acc00[v] + g0, 0.f) * w0 + fmaxf(acc10[v] + g1, 0.f) * w1;
    c1 += fmaxf(acc01[v] + g0, 0.f) * w0 + fmaxf(acc11[v] + g1, 0.f) * w1;
  }
  ldsPart[tid]       = c0;   // pos tile 0
  ldsPart[256 + tid] = c1;   // pos tile 1
  __syncthreads();

  if (tid < 32) {   // one thread per position: sum 8 waves x 2 lane-halves
    const int pt = tid >> 4, p = tid & 15;
    const float* part = ldsPart + pt * 256;
    float s = bf[0];
    #pragma unroll
    for (int w = 0; w < 8; ++w)
      s += part[w * 32 + p] + part[w * 32 + 16 + p];
    att_raw[(size_t)b * HW + posBase + pt * 16 + p] = s;
  }
}

// ---------------------------------------------------------------------------
// Kernel 3: per-batch  att = (att - min) / sum(att - min)
// ---------------------------------------------------------------------------
__global__ void norm_kernel(const float* __restrict__ att_raw,
                            float* __restrict__ attn, float* __restrict__ out_att) {
  __shared__ float smin[256];
  __shared__ float ssum[256];
  const int b = blockIdx.x, tid = threadIdx.x;
  const float* ar = att_raw + (size_t)b * HW;

  float mn = 3.402823466e38f, sm = 0.f;
  for (int i = tid; i < HW; i += 256) {
    const float v = ar[i];
    mn = fminf(mn, v);
    sm += v;
  }
  smin[tid] = mn; ssum[tid] = sm;
  __syncthreads();
  for (int s = 128; s > 0; s >>= 1) {
    if (tid < s) {
      smin[tid] = fminf(smin[tid], smin[tid + s]);
      ssum[tid] += ssum[tid + s];
    }
    __syncthreads();
  }
  const float m   = smin[0];
  const float inv = 1.f / (ssum[0] - (float)HW * m);   // sum(att - min)
  for (int i = tid; i < HW; i += 256) {
    const float v = (ar[i] - m) * inv;
    attn[(size_t)b * HW + i]    = v;
    out_att[(size_t)b * HW + i] = v;
  }
}

// ---------------------------------------------------------------------------
// Kernel 4: out[b,c] = sum_n x[b,c,n] * attn[b,n]   (HBM-bound, float4)
// ---------------------------------------------------------------------------
__global__ void pool_kernel(const float* __restrict__ x, const float* __restrict__ attn,
                            float* __restrict__ out) {
  __shared__ float red[256];
  const int b   = blockIdx.x >> 8;
  const int c   = blockIdx.x & 255;
  const int tid = threadIdx.x;
  const float4* xr = (const float4*)(x + ((size_t)b * Cin + c) * HW);
  const float4* ar = (const float4*)(attn + (size_t)b * HW);
  float acc = 0.f;
  for (int i = tid; i < HW / 4; i += 256) {
    const float4 xv = xr[i], av = ar[i];
    acc = fmaf(xv.x, av.x, acc);
    acc = fmaf(xv.y, av.y, acc);
    acc = fmaf(xv.z, av.z, acc);
    acc = fmaf(xv.w, av.w, acc);
  }
  red[tid] = acc;
  __syncthreads();
  for (int s = 128; s > 0; s >>= 1) {
    if (tid < s) red[tid] += red[tid + s];
    __syncthreads();
  }
  if (tid == 0) out[(size_t)b * Cin + c] = red[0];
}

// ---------------------------------------------------------------------------
extern "C" void kernel_launch(void* const* d_in, const int* in_sizes, int n_in,
                              void* d_out, int out_size, void* d_ws, size_t ws_size,
                              hipStream_t stream) {
  const float* x  = (const float*)d_in[0];  // [B, Cin, H, W]
  const float* g  = (const float*)d_in[1];  // [B, Cout]
  const float* Wg = (const float*)d_in[2];  // [A, Cout]
  const float* bg = (const float*)d_in[3];  // [A]
  const float* Wx = (const float*)d_in[4];  // [A, Cin]
  const float* bx = (const float*)d_in[5];  // [A]
  const float* Wf = (const float*)d_in[6];  // [1, A]
  const float* bf = (const float*)d_in[7];  // [1]

  float* out = (float*)d_out;                // [B*Cin] then [B*HW]
  float* ws  = (float*)d_ws;
  float*  ggbx   = ws;                                   //   8192 f
  float*  attraw = ggbx + Bsz * Adim;                    // 131072 f
  float*  attn   = attraw + (size_t)Bsz * HW;            // 131072 f
  __bf16* wxswz  = (__bf16*)(attn + (size_t)Bsz * HW);   //  65536 bf16 (128 KB)

  wx_swizzle_kernel<<<Adim * Cin / 256, 256, 0, stream>>>(Wx, wxswz);
  gate_kernel<<<Bsz, Adim, 0, stream>>>(g, Wg, bg, bx, ggbx);

  dim3 grid(HW / PT, Bsz);
  att_gemm_kernel<<<grid, 256, 0, stream>>>(x, wxswz, Wf, bf, ggbx, attraw);

  norm_kernel<<<Bsz, 256, 0, stream>>>(attraw, attn, out + Bsz * Cin);

  pool_kernel<<<Bsz * Cin, 256, 0, stream>>>(x, attn, out);
}
